// RecommenderPyTorch_4621384810498
// MI455X (gfx1250) — matrix operand order repaired and checked by reading, not compiled, not run on hardware
//
#include <hip/hip_runtime.h>

// ---------------------------------------------------------------------------
// CDNA5 (gfx1250) implementation. wave32, WMMA f32_16x16x32_f16, async LDS copy.
// ---------------------------------------------------------------------------

typedef __attribute__((ext_vector_type(16))) _Float16 v16h;
typedef __attribute__((ext_vector_type(8)))  _Float16 v8h;
typedef __attribute__((ext_vector_type(8)))  float    v8f;
typedef int v4i_vs __attribute__((vector_size(16)));   // matches builtin param type

#define LEAKY_S 0.5f

#if defined(__gfx1250__) && __has_builtin(__builtin_amdgcn_global_load_async_to_lds_b128) && __has_builtin(__builtin_amdgcn_s_wait_asynccnt)
#define USE_ASYNC_LDS 1
#else
#define USE_ASYNC_LDS 0
#endif

__device__ __forceinline__ float sigf(float x)   { return 1.f / (1.f + __expf(-x)); }
__device__ __forceinline__ float leakyf(float x) { return x >= 0.f ? x : LEAKY_S * x; }

// A-fragment (16x32 f16), rows of 64 halfs. Per ISA 7.12.2:
// lanes 0-15: row M=lane, h[0..7]=K k0..k0+7, h[8..15]=K k0+16..k0+23
// lanes 16-31: row M=lane-16, K offset +8 / +24.
__device__ __forceinline__ v16h frag_a(const _Float16* p, int kchunk) {
    const int lane = threadIdx.x & 31;
    const _Float16* r = p + (lane & 15) * 64 + kchunk * 32 + (lane >> 4) * 8;
    v8h lo = *(const v8h*)(r);
    v8h hh = *(const v8h*)(r + 16);
    v16h a;
#pragma unroll
    for (int i = 0; i < 8; i++) { a[i] = lo[i]; a[8 + i] = hh[i]; }
    return a;
}

// B-fragment (32x16 f16) from n-major [Ntot][64] f16 weight (pre-transposed;
// same K order as frag_a: k0+8h..+7 and k0+16+8h..+7 -> two 16B loads).
__device__ __forceinline__ v16h frag_b(const _Float16* W, int ntile, int kchunk) {
    const int lane = threadIdx.x & 31;
    const _Float16* r = W + (ntile * 16 + (lane & 15)) * 64 + kchunk * 32 + (lane >> 4) * 8;
    v8h lo = *(const v8h*)(r);
    v8h hh = *(const v8h*)(r + 16);
    v16h b;
#pragma unroll
    for (int i = 0; i < 8; i++) { b[i] = lo[i]; b[8 + i] = hh[i]; }
    return b;
}

__device__ __forceinline__ v8f wmma_f16(v16h a, v16h b, v8f c) {
    return __builtin_amdgcn_wmma_f32_16x16x32_f16(false, a, false, b, (short)0, c, false, false);
}

// ---------------------------------------------------------------------------
// Kernel 0: weight conversion / transposition to f16, bias sum.
// hws layout (halves): Wih[0,16384) Whh[16384,32768)
//   WqT0 32768  WkT0 36864  WvT0 40960
//   WqT1 45056  WkT1 49152  WvT1 53248
//   WvTseq0 57344  WvTseq1 61440    (total 65536 halves)
// ---------------------------------------------------------------------------
__global__ __launch_bounds__(256)
void convert_kernel(const float* __restrict__ Wih, const float* __restrict__ Whh,
                    const float* __restrict__ bih, const float* __restrict__ bhh,
                    const float* __restrict__ Wq0, const float* __restrict__ Wk0,
                    const float* __restrict__ Wv0, const float* __restrict__ Wq1,
                    const float* __restrict__ Wk1, const float* __restrict__ Wv1,
                    const float* __restrict__ Wv_seq,
                    _Float16* __restrict__ hws, float* __restrict__ bsum) {
    const int i = blockIdx.x * 256 + threadIdx.x;
    if (i < 16384) {
        hws[i]         = (_Float16)Wih[i];
        hws[16384 + i] = (_Float16)Whh[i];
    }
    if (i < 4096) {
        const int n = i >> 6, k = i & 63;
        const int t = k * 64 + n;                   // transpose: WT[n][k] = W[k][n]
        hws[32768 + i] = (_Float16)Wq0[t];
        hws[36864 + i] = (_Float16)Wk0[t];
        hws[40960 + i] = (_Float16)Wv0[t];
        hws[45056 + i] = (_Float16)Wq1[t];
        hws[49152 + i] = (_Float16)Wk1[t];
        hws[53248 + i] = (_Float16)Wv1[t];
        hws[57344 + i] = (_Float16)Wv_seq[t];
        hws[61440 + i] = (_Float16)Wv_seq[4096 + t];
    }
    if (i < 256) bsum[i] = bih[i] + bhh[i];
}

// ---------------------------------------------------------------------------
// Kernel 1: per-entity LSTM(4 steps) + LayerNorm + MHSA(T=4,H=4) + mean.
// One wave (32 threads) per 16-entity tile. 352 WMMAs / tile.
// LDS overlay keeps the block at 32 KB for ~10 workgroups / 320KB WGP.
// ---------------------------------------------------------------------------
__global__ __launch_bounds__(32)
void entity_wmma_kernel(const float* __restrict__ emb, int N,
                        const _Float16* __restrict__ Wih_h,
                        const _Float16* __restrict__ Whh_h,
                        const float* __restrict__ bsum,
                        const _Float16* __restrict__ WqT,
                        const _Float16* __restrict__ WkT,
                        const _Float16* __restrict__ WvT,
                        float* __restrict__ outv) {
    __shared__ __align__(16) _Float16     xbuf[4096];      // 8 KB
    __shared__ __align__(16) unsigned char region2[24576]; // 24 KB overlay:
    // prologue: float   stage[4096]            (16 KB)
    // LSTM:     float   hs[4][1024] (16 KB) + _Float16 hcur[1024] (2 KB)
    // MHSA:     _Float16 qkv[3][4096]          (24 KB)
    float*    const stage = (float*)region2;
    float*    const hs    = (float*)region2;
    _Float16* const hcur  = (_Float16*)(region2 + 16384);
    _Float16* const qkv   = (_Float16*)region2;

    const int lane = threadIdx.x;
    const int base = blockIdx.x * 16;
    const int hi   = lane >> 4;
    const int nlo  = lane & 15;

    // ---- stage x_t = 3 * emb[t, base+u, :] (t=0..3) into xbuf as f16 ----
#if USE_ASYNC_LDS
    {
        // 4096 f32 = 32 async b128 copies (32 lanes x 16B each): ASYNCcnt path.
#pragma unroll 4
        for (int inst = 0; inst < 32; ++inst) {
            const int f = inst * 128 + lane * 4;       // t*1024 + u*64 + d
            const int t = f >> 10, u = (f >> 6) & 15, d = f & 63;
            const float* gp = emb + ((size_t)t * N + (base + u)) * 64 + d;
            __builtin_amdgcn_global_load_async_to_lds_b128(
                (__attribute__((address_space(1))) v4i_vs*)(uintptr_t)gp,
                (__attribute__((address_space(3))) v4i_vs*)(stage + f), 0, 0);
        }
        __builtin_amdgcn_s_wait_asynccnt(0);
        __syncthreads();
        for (int i = lane; i < 4096; i += 32)
            xbuf[i] = (_Float16)(3.0f * stage[i]);     // same linearization
    }
#else
    for (int i = lane; i < 4096; i += 32) {
        const int t = i >> 10, rem = i & 1023;
        const int u = rem >> 6, d = rem & 63;
        const float* gp = emb + ((size_t)t * N + (base + u)) * 64 + d;
        xbuf[i] = (_Float16)(3.0f * __builtin_nontemporal_load(gp));
    }
#endif
    for (int i = lane; i < 1024; i += 32) hcur[i] = (_Float16)0.f;
    __syncthreads();

    // per-lane gate biases: bias[nt] for N = nt*16 + nlo
    float bias[16];
#pragma unroll
    for (int nt = 0; nt < 16; nt++) bias[nt] = bsum[nt * 16 + nlo];

    float creg[4][8];
#pragma unroll
    for (int dt = 0; dt < 4; dt++)
#pragma unroll
        for (int r = 0; r < 8; r++) creg[dt][r] = 0.f;

    // ---- LSTM: 4 time steps, gates = x_t@Wih^T + h@Whh^T + b  (16 x 256) ----
    for (int t = 0; t < 4; t++) {
        v16h ax0 = frag_a(xbuf + t * 1024, 0);
        v16h ax1 = frag_a(xbuf + t * 1024, 1);
        v16h ah0 = frag_a(hcur, 0);
        v16h ah1 = frag_a(hcur, 1);

        for (int dt = 0; dt < 4; dt++) {           // dim-tile (16 dims of 64)
            // preload ALL 16 B-fragments first -> ~32 b128 loads in flight
            v16h bW[16];
#pragma unroll
            for (int gate = 0; gate < 4; gate++) {
                const int nt = gate * 4 + dt;
                bW[gate * 4 + 0] = frag_b(Wih_h, nt, 0);
                bW[gate * 4 + 1] = frag_b(Wih_h, nt, 1);
                bW[gate * 4 + 2] = frag_b(Whh_h, nt, 0);
                bW[gate * 4 + 3] = frag_b(Whh_h, nt, 1);
            }
            v8f acc[4];                            // i,f,g,o accumulators
#pragma unroll
            for (int gate = 0; gate < 4; gate++) {
                const float bv = bias[gate * 4 + dt];
                v8f c;
#pragma unroll
                for (int r = 0; r < 8; r++) c[r] = bv;
                c = wmma_f16(ax0, bW[gate * 4 + 0], c);
                c = wmma_f16(ax1, bW[gate * 4 + 1], c);
                c = wmma_f16(ah0, bW[gate * 4 + 2], c);
                c = wmma_f16(ah1, bW[gate * 4 + 3], c);
                acc[gate] = c;
            }
            // LSTM cell update; (M,N) = (r + hi*8, nlo) per C/D VGPR layout
#pragma unroll
            for (int r = 0; r < 8; r++) {
                const float iv = sigf(acc[0][r]);
                const float fv = sigf(acc[1][r]);
                const float gv = tanhf(acc[2][r]);
                const float ov = sigf(acc[3][r]);
                const float cc = fv * creg[dt][r] + iv * gv;
                creg[dt][r] = cc;
                const float hh = ov * tanhf(cc);
                const int M = r + hi * 8;
                const int d = dt * 16 + nlo;
                hs[t * 1024 + M * 64 + d] = hh;
                hcur[M * 64 + d] = (_Float16)hh;
            }
        }
        __syncthreads();
    }

    // ---- LayerNorm(hs) -> xbuf, row order u*4+t (token-major for MHSA) ----
    for (int rr = lane; rr < 64; rr += 32) {
        const int u = rr >> 2, t = rr & 3;
        const float* hp = hs + t * 1024 + u * 64;
        float m = 0.f;
        for (int d = 0; d < 64; d++) m += hp[d];
        m *= (1.f / 64.f);
        float v = 0.f;
        for (int d = 0; d < 64; d++) { const float x = hp[d] - m; v += x * x; }
        const float rs = rsqrtf(v * (1.f / 64.f) + 1e-5f);
        for (int d = 0; d < 64; d++) xbuf[rr * 64 + d] = (_Float16)((hp[d] - m) * rs);
    }
    __syncthreads();   // hs dead from here; region2 becomes qkv

    // ---- Q/K/V projections: (64x64) @ (64x64); B-frags hoisted (4x reuse) ----
    const _Float16* const Ws[3] = { WqT, WkT, WvT };
#pragma unroll
    for (int w = 0; w < 3; w++) {
        v16h bf[8];
#pragma unroll
        for (int nt = 0; nt < 4; nt++) {
            bf[nt * 2 + 0] = frag_b(Ws[w], nt, 0);
            bf[nt * 2 + 1] = frag_b(Ws[w], nt, 1);
        }
        for (int mt = 0; mt < 4; mt++) {
            v16h a0 = frag_a(xbuf + mt * 1024, 0);
            v16h a1 = frag_a(xbuf + mt * 1024, 1);
#pragma unroll
            for (int nt = 0; nt < 4; nt++) {
                v8f c;
#pragma unroll
                for (int r = 0; r < 8; r++) c[r] = 0.f;
                c = wmma_f16(a0, bf[nt * 2 + 0], c);
                c = wmma_f16(a1, bf[nt * 2 + 1], c);
#pragma unroll
                for (int r = 0; r < 8; r++) {
                    const int row = mt * 16 + r + hi * 8;
                    const int col = nt * 16 + nlo;
                    qkv[w * 4096 + row * 64 + col] = (_Float16)c[r];
                }
            }
        }
    }
    __syncthreads();

    // ---- per-(user,head) attention over T=4, then mean over T ----
    for (int unit = lane; unit < 64; unit += 32) {
        const int u = unit >> 2, h = unit & 3;
        const _Float16* qp = qkv + 0 * 4096 + (u * 4) * 64 + h * 16;
        const _Float16* kp = qkv + 1 * 4096 + (u * 4) * 64 + h * 16;
        const _Float16* vp = qkv + 2 * 4096 + (u * 4) * 64 + h * 16;
        float accd[16];
#pragma unroll
        for (int d = 0; d < 16; d++) accd[d] = 0.f;
        for (int t = 0; t < 4; t++) {
            float s[4];
#pragma unroll
            for (int ss = 0; ss < 4; ss++) {
                float dot = 0.f;
                for (int d = 0; d < 16; d++) dot += (float)qp[t * 64 + d] * (float)kp[ss * 64 + d];
                s[ss] = dot * 0.25f;               // 1/sqrt(dk=16)
            }
            const float mx = fmaxf(fmaxf(s[0], s[1]), fmaxf(s[2], s[3]));
            float e[4], se = 0.f;
#pragma unroll
            for (int ss = 0; ss < 4; ss++) { e[ss] = __expf(s[ss] - mx); se += e[ss]; }
            const float inv = 1.f / se;
#pragma unroll
            for (int ss = 0; ss < 4; ss++) {
                const float a = e[ss] * inv;
                for (int d = 0; d < 16; d++) accd[d] += a * (float)vp[ss * 64 + d];
            }
        }
        float* op = outv + (size_t)(base + u) * 64 + h * 16;
        for (int d = 0; d < 16; d++) op[d] = accd[d] * 0.25f;  // mean over T=4
    }
}

// ---------------------------------------------------------------------------
// Kernel 2: sequence branch. T=1 -> mhsa(x) == x @ Wv_seq[i].
// ---------------------------------------------------------------------------
__global__ __launch_bounds__(32)
void seq_att_kernel(const float* __restrict__ final_item,
                    const float* __restrict__ posEmbed,
                    const float* __restrict__ mask,
                    const int* __restrict__ sequence,
                    const _Float16* __restrict__ WvTseq,   // [2][64*64]
                    float* __restrict__ att_user) {
    __shared__ __align__(16) float    attb[16 * 64];
    __shared__ __align__(16) _Float16 xh[16 * 64];
    const int lane = threadIdx.x;
    const int hi = lane >> 4, nlo = lane & 15;
    const int b0 = blockIdx.x * 16;

    // sb = LN( sum_l mask * (final_item[seq] + pos) )
    if (lane < 16) {
        const int b = b0 + lane;
        float acc[64];
        for (int d = 0; d < 64; d++) acc[d] = 0.f;
        for (int l = 0; l < 50; l++) {
            const float mm = mask[b * 50 + l];
            const int id   = sequence[b * 50 + l];
            const float* fi = final_item + (size_t)id * 64;
            const float* pe = posEmbed + l * 64;
            for (int d = 0; d < 64; d++) acc[d] += mm * (fi[d] + pe[d]);
        }
        float m = 0.f;
        for (int d = 0; d < 64; d++) m += acc[d];
        m *= (1.f / 64.f);
        float v = 0.f;
        for (int d = 0; d < 64; d++) { const float x = acc[d] - m; v += x * x; }
        const float rs = rsqrtf(v * (1.f / 64.f) + 1e-5f);
        for (int d = 0; d < 64; d++) attb[lane * 64 + d] = (acc[d] - m) * rs;
    }
    __syncthreads();

    for (int li = 0; li < 2; li++) {
        if (lane < 16) {                       // LN(att) -> f16 A operand
            const float* ap = &attb[lane * 64];
            float m = 0.f;
            for (int d = 0; d < 64; d++) m += ap[d];
            m *= (1.f / 64.f);
            float v = 0.f;
            for (int d = 0; d < 64; d++) { const float x = ap[d] - m; v += x * x; }
            const float rs = rsqrtf(v * (1.f / 64.f) + 1e-5f);
            for (int d = 0; d < 64; d++) xh[lane * 64 + d] = (_Float16)((ap[d] - m) * rs);
        }
        __syncthreads();
        v16h a0 = frag_a(xh, 0);
        v16h a1 = frag_a(xh, 1);
        const _Float16* W = WvTseq + li * 4096;
        for (int nt = 0; nt < 4; nt++) {
            v8f c;
#pragma unroll
            for (int r = 0; r < 8; r++) c[r] = 0.f;
            c = wmma_f16(a0, frag_b(W, nt, 0), c);
            c = wmma_f16(a1, frag_b(W, nt, 1), c);
#pragma unroll
            for (int r = 0; r < 8; r++) {
                const int M = r + hi * 8, d = nt * 16 + nlo;
                attb[M * 64 + d] += leakyf(c[r]);   // att = leaky(a1) + att
            }
        }
        __syncthreads();
    }
    for (int i = lane; i < 1024; i += 32)
        att_user[(size_t)b0 * 64 + i] = attb[i];
}

// ---------------------------------------------------------------------------
// Kernel 3: preds[p] = <fu[uids], fi[iids]> + <leaky(att_user[uloc]), fi[iids]>
// ---------------------------------------------------------------------------
__global__ __launch_bounds__(256)
void preds_kernel(const float* __restrict__ final_user,
                  const float* __restrict__ final_item,
                  const float* __restrict__ att_user,
                  const int* __restrict__ uids, const int* __restrict__ iids,
                  const int* __restrict__ uloc, float* __restrict__ out) {
    const int p = blockIdx.x * 256 + threadIdx.x;
    const float* fu = final_user + (size_t)uids[p] * 64;
    const float* fi = final_item + (size_t)iids[p] * 64;
    const float* au = att_user  + (size_t)uloc[p] * 64;
    float s = 0.f;
    for (int d = 0; d < 64; d++) s += (fu[d] + leakyf(au[d])) * fi[d];
    out[p] = s;
}

// ---------------------------------------------------------------------------
// Kernel 4: meta chain / S_final / p1 only at gathered (g, suids/siids):
// user_weight is only read at suids, so skip the 4x100K dense meta GEMM.
// ---------------------------------------------------------------------------
__global__ __launch_bounds__(256)
void meta_kernel(const float* __restrict__ final_user,
                 const float* __restrict__ final_item,
                 const float* __restrict__ uEmbed, const float* __restrict__ iEmbed,
                 const int* __restrict__ suids, const int* __restrict__ siids,
                 const float* __restrict__ m2W, const float* __restrict__ m2b,
                 const float* __restrict__ m3W, const float* __restrict__ m3b,
                 float* __restrict__ uw, float* __restrict__ sfin,
                 float* __restrict__ p1w) {
    const int e = blockIdx.x * 256 + threadIdx.x;     // 0..16383 = g*4096 + j
    const int g = e >> 12;
    const int u = suids[e], it = siids[e];
    const float* fu = final_user + (size_t)u * 64;
    const float* fi = final_item + (size_t)it * 64;
    const float* up = uEmbed + ((size_t)g * 100000 + u) * 64;
    const float* ip = iEmbed + ((size_t)g * 50000 + it) * 64;
    float m1[192];
    float sf = 0.f, pp = 0.f;
    for (int d = 0; d < 64; d++) {
        const float uv = 3.f * up[d], iv = 3.f * ip[d], fud = fu[d];
        m1[d] = fud * uv; m1[64 + d] = fud; m1[128 + d] = uv;
        sf += leakyf(fud * fi[d]);
        pp += leakyf(uv * iv);
    }
    float a3 = m3b[0];
    for (int o = 0; o < 32; o++) {
        const float* wr = m2W + o * 192;
        float a = m2b[o];
        for (int d = 0; d < 192; d++) a += m1[d] * wr[d];
        a3 += leakyf(a) * m3W[o];
    }
    uw[e] = sigf(a3);
    sfin[e] = sf;
    p1w[e] = pp;
}

// ---------------------------------------------------------------------------
// Kernel 5/6: deterministic two-stage SSL hinge reduction.
// ---------------------------------------------------------------------------
__global__ __launch_bounds__(256)
void ssl_partial_kernel(const float* __restrict__ uw, const float* __restrict__ sfin,
                        const float* __restrict__ p1w, float* __restrict__ partials) {
    __shared__ float red[256];
    const int e = blockIdx.x * 256 + threadIdx.x;     // 0..8191 = g*2048 + s
    const int g = e >> 11, s = e & 2047;
    const int jp = g * 4096 + s, jn = jp + 2048;
    const float Sf = uw[jp] * sfin[jp] - uw[jn] * sfin[jn];
    const float diff = p1w[jp] - p1w[jn];
    red[threadIdx.x] = fmaxf(1.f - Sf * diff, 0.f);
    __syncthreads();
    for (int off = 128; off > 0; off >>= 1) {
        if (threadIdx.x < off) red[threadIdx.x] += red[threadIdx.x + off];
        __syncthreads();
    }
    if (threadIdx.x == 0) partials[blockIdx.x] = red[0];
}

__global__ void ssl_final_kernel(const float* __restrict__ partials, float* __restrict__ out) {
    if (threadIdx.x == 0) {
        float s = 0.f;
        for (int i = 0; i < 32; i++) s += partials[i];
        *out = s;
    }
}

// ---------------------------------------------------------------------------
extern "C" void kernel_launch(void* const* d_in, const int* in_sizes, int n_in,
                              void* d_out, int out_size, void* d_ws, size_t ws_size,
                              hipStream_t stream) {
    (void)in_sizes; (void)n_in; (void)out_size; (void)ws_size;
    const float* uEmbed   = (const float*)d_in[0];
    const float* iEmbed   = (const float*)d_in[1];
    const float* posEmbed = (const float*)d_in[2];
    const float* Wih      = (const float*)d_in[3];
    const float* Whh      = (const float*)d_in[4];
    const float* bih      = (const float*)d_in[5];
    const float* bhh      = (const float*)d_in[6];
    const float* Wq0      = (const float*)d_in[7];
    const float* Wk0      = (const float*)d_in[8];
    const float* Wv0      = (const float*)d_in[9];
    const float* Wq1      = (const float*)d_in[10];
    const float* Wk1      = (const float*)d_in[11];
    const float* Wv1      = (const float*)d_in[12];
    const float* Wv_seq   = (const float*)d_in[15];
    const float* meta2_W  = (const float*)d_in[16];
    const float* meta2_b  = (const float*)d_in[17];
    const float* meta3_W  = (const float*)d_in[18];
    const float* meta3_b  = (const float*)d_in[19];
    const float* mask     = (const float*)d_in[20];
    const int*   uids     = (const int*)d_in[21];
    const int*   iids     = (const int*)d_in[22];
    const int*   sequence = (const int*)d_in[23];
    const int*   uLocs    = (const int*)d_in[24];
    const int*   suids    = (const int*)d_in[25];
    const int*   siids    = (const int*)d_in[26];

    // workspace layout (all 16B aligned)
    float* ws         = (float*)d_ws;
    float* final_user = ws;                          // 6,400,000 f
    float* final_item = final_user + 6400000;        // 3,200,000 f
    float* att_user   = final_item + 3200000;        // 131,072 f
    float* uw         = att_user + 131072;           // 16,384 f
    float* sfin       = uw + 16384;                  // 16,384 f
    float* p1w        = sfin + 16384;                // 16,384 f
    float* partials   = p1w + 16384;                 // 64 f
    float* bsum       = partials + 64;               // 256 f
    _Float16* hws     = (_Float16*)(bsum + 256);     // 65,536 halves

    convert_kernel<<<64, 256, 0, stream>>>(Wih, Whh, bih, bhh, Wq0, Wk0, Wv0,
                                           Wq1, Wk1, Wv1, Wv_seq, hws, bsum);

    entity_wmma_kernel<<<6250, 32, 0, stream>>>(uEmbed, 100000, hws, hws + 16384, bsum,
                                                hws + 32768, hws + 36864, hws + 40960,
                                                final_user);
    entity_wmma_kernel<<<3125, 32, 0, stream>>>(iEmbed, 50000, hws, hws + 16384, bsum,
                                                hws + 45056, hws + 49152, hws + 53248,
                                                final_item);

    seq_att_kernel<<<128, 32, 0, stream>>>(final_item, posEmbed, mask, sequence,
                                           hws + 57344, att_user);

    preds_kernel<<<32, 256, 0, stream>>>(final_user, final_item, att_user,
                                         uids, iids, uLocs, (float*)d_out);

    meta_kernel<<<64, 256, 0, stream>>>(final_user, final_item, uEmbed, iEmbed,
                                        suids, siids, meta2_W, meta2_b, meta3_W, meta3_b,
                                        uw, sfin, p1w);

    ssl_partial_kernel<<<32, 256, 0, stream>>>(uw, sfin, p1w, partials);
    ssl_final_kernel<<<1, 32, 0, stream>>>(partials, (float*)d_out + 8192);
}
